// RCCA_76459007804004
// MI455X (gfx1250) — matrix operand order repaired and checked by reading, not compiled, not run on hardware
//
#include <hip/hip_runtime.h>
#include <hip/hip_bf16.h>

typedef __attribute__((ext_vector_type(16))) _Float16 v16h;
typedef __attribute__((ext_vector_type(8)))  _Float16 v8h;
typedef __attribute__((ext_vector_type(8)))  float    v8f;
typedef __attribute__((ext_vector_type(4)))  float    v4f;

#define WMMA_F16(A, B, C) \
  __builtin_amdgcn_wmma_f32_16x16x32_f16(false, (A), false, (B), (short)0, (C), false, false)

constexpr int Bsz = 4;
constexpr int Cch = 256;   // channels
constexpr int Nsq = 4096;  // sequence positions

// ---------------------------------------------------------------------------
// Kernel 0: convert the three 256x256 weight matrices to f16 (packed [3][C][C])
// ---------------------------------------------------------------------------
__global__ void wcvt_kernel(const float* __restrict__ wq,
                            const float* __restrict__ wk,
                            const float* __restrict__ wv,
                            _Float16* __restrict__ out) {
  int t = blockIdx.x * 256 + threadIdx.x;       // 0 .. 3*65536-1
  int m = t >> 16, e = t & 65535;
  const float* s = (m == 0) ? wq : (m == 1) ? wk : wv;
  out[t] = (_Float16)s[e];
}

// ---------------------------------------------------------------------------
// Kernel 1: QKV projection via WMMA.
//   Qt, Kt : [B][N][C] f16  (row n, contiguous channels -> A/B frags are b128)
//   V      : [B][C][N] f16  (row c, contiguous j        -> B frags are b128)
// One wave handles a 16-row n tile; x tile transposed through LDS once.
// ---------------------------------------------------------------------------
__global__ void __launch_bounds__(128, 1)
qkv_kernel(const float* __restrict__ x,         // [B][C][N] f32
           const _Float16* __restrict__ wh,     // [3][C][C] f16
           const float* __restrict__ bq,
           const float* __restrict__ bk,
           const float* __restrict__ bv,
           _Float16* __restrict__ Qt,
           _Float16* __restrict__ Kt,
           _Float16* __restrict__ V) {
  __shared__ _Float16 xT[4][16 * 256];          // per-wave x^T tile (8KB each)
  const int wave = threadIdx.x >> 5, lane = threadIdx.x & 31;
  const int tile = blockIdx.x * 4 + wave;       // 0..1023
  const int b  = tile >> 8;                     // 256 n-tiles per batch
  const int n0 = (tile & 255) << 4;
  const int row = lane & 15, half = lane >> 4;

  _Float16* xt = xT[wave];
  const float* xb = x + (size_t)b * Cch * Nsq;
  // stage x[b][:, n0..n0+15] transposed into LDS as xT[n][c] f16
  #pragma unroll 4
  for (int it = 0; it < 128; ++it) {
    int idx = it * 32 + lane;
    int c = idx >> 4, n = idx & 15;
    xt[n * 256 + c] = (_Float16)xb[(size_t)c * Nsq + n0 + n];
  }
  asm volatile("s_wait_dscnt 0x0" ::: "memory");

  // A fragments of x^T (M=n, K=c), K=256 in 8 chunks of 32
  v16h a[8];
  #pragma unroll
  for (int k = 0; k < 8; ++k) {
    ((v8h*)&a[k])[0] = *(const v8h*)&xt[row * 256 + k * 32 + half * 16];
    ((v8h*)&a[k])[1] = *(const v8h*)&xt[row * 256 + k * 32 + half * 16 + 8];
  }

  #pragma unroll
  for (int m = 0; m < 3; ++m) {                 // m constant after unroll:
    const _Float16* W = wh + (size_t)m * 65536; //   no flat loads, no branches
    const float* bias = (m == 0) ? bq : (m == 1) ? bk : bv;
    for (int ot = 0; ot < 16; ++ot) {
      const int o = ot * 16 + row;              // this lane's output column
      v8f acc = {};
      // 2-deep pipelined B-fragment loads (K=c, N=o)
      v16h wa = *(const v16h*)&W[o * 256 + half * 16];
      #pragma unroll
      for (int k = 0; k < 8; ++k) {
        v16h wn = wa;
        if (k < 7) wn = *(const v16h*)&W[o * 256 + (k + 1) * 32 + half * 16];
        acc = WMMA_F16(a[k], wa, acc);
        wa = wn;
      }
      const float bo = bias[o];
      #pragma unroll
      for (int v = 0; v < 8; ++v) {             // D: (M = v + 8*half, N = row)
        float val = acc[v] + bo;
        int n = n0 + v + half * 8;
        if (m == 0)      Qt[((size_t)b * Nsq + n) * Cch + o] = (_Float16)val;
        else if (m == 1) Kt[((size_t)b * Nsq + n) * Cch + o] = (_Float16)val;
        else             V[((size_t)b * Cch + o) * (size_t)Nsq + n] = (_Float16)val;
      }
    }
  }
}

// ---------------------------------------------------------------------------
// Kernel 2: flash attention + residual. One wave = 16 query rows x 256 channels.
// ---------------------------------------------------------------------------
__global__ void __launch_bounds__(128, 1)
attn_kernel(const float* __restrict__ x,
            const _Float16* __restrict__ Qt,   // [B][N][C]
            const _Float16* __restrict__ Kt,   // [B][N][C]
            const _Float16* __restrict__ Vm,   // [B][C][N]
            float* __restrict__ out) {
  __shared__ _Float16 pbuf[4][16 * 32];        // per-wave prob tile (1KB each)
  __shared__ float    cbuf[4][128 * 16];       // per-wave epilogue tile (8KB)
  const int wave = threadIdx.x >> 5, lane = threadIdx.x & 31;
  const int tile = blockIdx.x * 4 + wave;      // 0..1023
  const int b  = tile >> 8;
  const int i0 = (tile & 255) << 4;
  const int row = lane & 15, half = lane >> 4;
  const float C1 = 0.0625f * 1.44269504089f;   // (1/sqrt(256)) * log2(e)

  // Q A-fragments (M=i, K=c), held for the whole sweep: 64 VGPRs
  v16h q[8];
  const _Float16* Qb = Qt + ((size_t)b * Nsq + i0 + row) * Cch;
  #pragma unroll
  for (int k = 0; k < 8; ++k) {
    ((v8h*)&q[k])[0] = *(const v8h*)&Qb[k * 32 + half * 16];
    ((v8h*)&q[k])[1] = *(const v8h*)&Qb[k * 32 + half * 16 + 8];
  }

  v8f acc[16];                                  // context^T tile [16 i][256 c]
  #pragma unroll
  for (int ct = 0; ct < 16; ++ct) { v8f z = {}; acc[ct] = z; }
  float mrow[8], srow[8];                       // online-softmax stats per row
  #pragma unroll
  for (int v = 0; v < 8; ++v) { mrow[v] = -3.0e38f; srow[v] = 0.f; }
  _Float16* pl = pbuf[wave];

  for (int j0 = 0; j0 < Nsq; j0 += 32) {
    // ---- scores: two 16x16 tiles (j0..15, j0+16..31), K = 256 ----
    const _Float16* K0 = Kt + ((size_t)b * Nsq + j0 + row) * Cch + half * 16;
    const _Float16* K1 = K0 + 16 * Cch;
    const _Float16* Vb = Vm + ((size_t)b * Cch + row) * (size_t)Nsq + j0 + half * 16;
    __builtin_prefetch(K0 + 32 * Cch, 0, 1);    // next j-tile of K
    __builtin_prefetch(Vb + 32, 0, 1);          // next j-tile of V
    v8f sD0 = {}, sD1 = {};
    {   // 2-deep pipelined K-fragment loads
      v16h ka0 = *(const v16h*)&K0[0];
      v16h ka1 = *(const v16h*)&K1[0];
      #pragma unroll
      for (int k = 0; k < 8; ++k) {
        v16h kb0 = ka0, kb1 = ka1;
        if (k < 7) {
          kb0 = *(const v16h*)&K0[(k + 1) * 32];
          kb1 = *(const v16h*)&K1[(k + 1) * 32];
        }
        sD0 = WMMA_F16(q[k], ka0, sD0);
        sD1 = WMMA_F16(q[k], ka1, sD1);
        ka0 = kb0; ka1 = kb1;
      }
    }
    // ---- online softmax, one row (VGPR index v) at a time ----
    #pragma unroll
    for (int v = 0; v < 8; ++v) {
      float lm = fmaxf(sD0[v], sD1[v]);         // row-wise max across N lanes
      lm = fmaxf(lm, __shfl_xor(lm, 8, 32));
      lm = fmaxf(lm, __shfl_xor(lm, 4, 32));
      lm = fmaxf(lm, __shfl_xor(lm, 2, 32));
      lm = fmaxf(lm, __shfl_xor(lm, 1, 32));
      const float mn  = fmaxf(mrow[v], lm);
      const float cor = __builtin_exp2f((mrow[v] - mn) * C1);
      mrow[v] = mn;
      const float p0 = __builtin_exp2f((sD0[v] - mn) * C1);
      const float p1 = __builtin_exp2f((sD1[v] - mn) * C1);
      float rs = p0 + p1;
      rs += __shfl_xor(rs, 8, 32);
      rs += __shfl_xor(rs, 4, 32);
      rs += __shfl_xor(rs, 2, 32);
      rs += __shfl_xor(rs, 1, 32);
      srow[v] = srow[v] * cor + rs;
      const int ir = v + half * 8;              // this element's query row
      pl[ir * 32 + row]      = (_Float16)p0;    // prob tile [i][j] in LDS
      pl[ir * 32 + 16 + row] = (_Float16)p1;
      #pragma unroll
      for (int ct = 0; ct < 16; ++ct) acc[ct][v] *= cor;  // rescale context
    }
    asm volatile("s_wait_dscnt 0x0" ::: "memory");
    // ---- re-layout probs as A fragment (M=i, K=j, 16x32) from LDS ----
    v16h pa;
    ((v8h*)&pa)[0] = *(const v8h*)&pl[row * 32 + half * 16];
    ((v8h*)&pa)[1] = *(const v8h*)&pl[row * 32 + half * 16 + 8];
    // ---- context += attn x V^T : 16 channel tiles, 2-deep pipelined ----
    {
      v16h va = *(const v16h*)&Vb[0];
      #pragma unroll
      for (int ct = 0; ct < 16; ++ct) {
        v16h vn = va;
        if (ct < 15) vn = *(const v16h*)&Vb[(size_t)(ct + 1) * 16 * Nsq];
        acc[ct] = WMMA_F16(pa, va, acc[ct]);
        va = vn;
      }
    }
  }

  // ---- epilogue: out = x + context / rowsum.
  // Transpose context through LDS (two 128-channel passes) so the residual
  // add runs on full 64-byte channel rows with b128 loads/stores.
  float rinv[8];
  #pragma unroll
  for (int v = 0; v < 8; ++v) rinv[v] = 1.0f / srow[v];
  float* cb = cbuf[wave];
  #pragma unroll
  for (int pass = 0; pass < 2; ++pass) {
    #pragma unroll
    for (int ct = 0; ct < 8; ++ct) {
      const int cl = ct * 16 + row;             // local channel 0..127
      #pragma unroll
      for (int v = 0; v < 8; ++v)
        cb[cl * 16 + v + half * 8] = acc[pass * 8 + ct][v] * rinv[v];
    }
    asm volatile("s_wait_dscnt 0x0" ::: "memory");
    #pragma unroll
    for (int r = 0; r < 4; ++r) {
      const int cl = r * 32 + lane;             // local channel 0..127
      const int c  = pass * 128 + cl;
      const size_t base = ((size_t)b * Cch + c) * (size_t)Nsq + i0;
      const v4f* xr = (const v4f*)&x[base];
      v4f*       orow = (v4f*)&out[base];
      #pragma unroll
      for (int ch = 0; ch < 4; ++ch) {
        v4f cv = *(const v4f*)&cb[cl * 16 + ch * 4];
        orow[ch] = xr[ch] + cv;
      }
    }
    asm volatile("s_wait_dscnt 0x0" ::: "memory");  // LDS reuse across passes
  }
}

// ---------------------------------------------------------------------------
extern "C" void kernel_launch(void* const* d_in, const int* in_sizes, int n_in,
                              void* d_out, int out_size, void* d_ws, size_t ws_size,
                              hipStream_t stream) {
  const float* x  = (const float*)d_in[0];
  const float* Wq = (const float*)d_in[1];
  const float* bq = (const float*)d_in[2];
  const float* Wk = (const float*)d_in[3];
  const float* bk = (const float*)d_in[4];
  const float* Wv = (const float*)d_in[5];
  const float* bv = (const float*)d_in[6];
  float* out = (float*)d_out;

  // workspace layout: [3][C][C] f16 weights | Qt | Kt | V  (~24.4 MiB total)
  char* ws = (char*)d_ws;
  _Float16* wh = (_Float16*)ws;
  _Float16* Qt = (_Float16*)(ws + (size_t)3 * Cch * Cch * 2);
  _Float16* Kt = Qt + (size_t)Bsz * Nsq * Cch;
  _Float16* V  = Kt + (size_t)Bsz * Nsq * Cch;

  wcvt_kernel<<<(3 * Cch * Cch) / 256, 256, 0, stream>>>(Wq, Wk, Wv, wh);
  // 1024 wave-tiles, 4 waves (128 threads) per block
  qkv_kernel<<<256, 128, 0, stream>>>(x, wh, bq, bk, bv, Qt, Kt, V);
  attn_kernel<<<256, 128, 0, stream>>>(x, Qt, Kt, V, out);
}
